// MultiHeadSelfAttention_40381282517603
// MI455X (gfx1250) — matrix operand order repaired and checked by reading, not compiled, not run on hardware
//
#include <hip/hip_runtime.h>
#include <cstdint>

// ---------------------------------------------------------------------------
// MI455X (gfx1250) multi-head self-attention, f16 WMMA pipeline:
//   x,w -> f16 (weights transposed) -> qkv GEMM -> flash attention
//   -> proj GEMM (+bias) -> f32 out
// Matrix math: v_wmma_f32_16x16x32_f16 (wave32).
// LDS staging: Tensor Data Mover (tensor_load_to_lds) with double buffering,
// tracked by TENSORcnt, so DMA overlaps WMMA issue.
// ---------------------------------------------------------------------------

typedef _Float16 half_t;
typedef __attribute__((ext_vector_type(8)))  _Float16 v8h;
typedef __attribute__((ext_vector_type(16))) _Float16 v16h;
typedef __attribute__((ext_vector_type(8)))  float    v8f;
typedef __attribute__((ext_vector_type(4)))  unsigned int u32x4;
typedef __attribute__((ext_vector_type(8)))  int      i32x8;
typedef __attribute__((ext_vector_type(4)))  int      i32x4;

#define WMMA_F32_F16(a, b, c) \
  __builtin_amdgcn_wmma_f32_16x16x32_f16(false, (a), false, (b), (short)0, (c), false, false)

static __device__ inline v16h cat8(v8h lo, v8h hi) {
  union { v16h v; v8h h[2]; } u;
  u.h[0] = lo; u.h[1] = hi;
  return u.v;
}

// Low 32 bits of a generic pointer into LDS = byte offset within the
// workgroup's LDS allocation (LDS aperture keeps the offset in addr[31:0]).
static __device__ inline uint32_t lds_lo32(const void* p) {
  return (uint32_t)(size_t)p;
}

// ---------------------------------------------------------------------------
// TDM: issue one tensor_load_to_lds for a 2D tile of 16-bit elements.
//   tile_w   : tile width in halves (tile_dim0, multiple of 2 -> 4 bytes)
//   tile_h   : tile height in rows  (tile_dim1)
//   stride_h : row stride in halves (tensor_dim0_stride)
// Descriptor per CDNA5 ISA (2D, groups 2/3 zero).
// Per-wave scalar op (EXEC ignored); caller gates to one wave per workgroup.
// ---------------------------------------------------------------------------
static __device__ inline void tdm_load_2d(uint32_t lds_addr, const half_t* gsrc,
                                          uint32_t tile_w, uint32_t tile_h,
                                          uint32_t stride_h) {
  uint64_t ga = (uint64_t)(size_t)gsrc;
  u32x4 g0;
  g0.x = 1u;                                              // count=1 (valid, user mode)
  g0.y = lds_addr;                                        // lds_addr[31:0]
  g0.z = (uint32_t)ga;                                    // global_addr[31:0]
  g0.w = (uint32_t)((ga >> 32) & 0x1FFFFFFu) | (2u << 30);// global_addr[56:32], type=2

  i32x8 g1;
  g1[0] = (int)(1u << 16);                                // wg_mask=0, data_size=1 (2B)
  g1[1] = (int)((tile_w & 0xFFFFu) << 16);                // tensor_dim0[15:0] (== tile_w)
  g1[2] = (int)(((tile_w >> 16) & 0xFFFFu) |              // tensor_dim0[31:16]
                ((tile_h & 0xFFFFu) << 16));              // tensor_dim1[15:0] (== tile_h)
  g1[3] = (int)(((tile_h >> 16) & 0xFFFFu) |              // tensor_dim1[31:16]
                ((tile_w & 0xFFFFu) << 16));              // tile_dim0
  g1[4] = (int)(tile_h & 0xFFFFu);                        // tile_dim1, tile_dim2=0
  g1[5] = (int)stride_h;                                  // tensor_dim0_stride[31:0]
  g1[6] = 0;                                              // stride0 hi, dim1_stride lo
  g1[7] = 0;                                              // dim1_stride hi

  i32x4 z4 = (i32x4)0;
#if defined(__clang_major__) && (__clang_major__ >= 23)
  i32x8 z8 = (i32x8)0;
  __builtin_amdgcn_tensor_load_to_lds(g0, g1, z4, z4, z8, 0);
#else
  __builtin_amdgcn_tensor_load_to_lds(g0, g1, z4, z4, 0);
#endif
}

// ---------------------------------------------------------------------------
// Precision conversion kernels
// ---------------------------------------------------------------------------
__global__ __launch_bounds__(256) void cvt_f32_to_f16(const float* __restrict__ src,
                                                      half_t* __restrict__ dst, int n) {
  int i = blockIdx.x * blockDim.x + threadIdx.x;
  if (i < n) dst[i] = (half_t)src[i];
}

// src is [K][N] row-major f32; dst is [N][K] row-major f16 (i.e. B^T).
__global__ __launch_bounds__(256) void transpose_f32_to_f16(const float* __restrict__ src,
                                                            half_t* __restrict__ dst,
                                                            int K, int N) {
  int i = blockIdx.x * blockDim.x + threadIdx.x;
  if (i >= K * N) return;
  int n = i / K;
  int k = i - n * K;
  dst[i] = (half_t)src[k * N + n];
}

// ---------------------------------------------------------------------------
// Shared GEMM tile core:  C[128x128] tile of A[M,K] x B[K,Ncols],
// with Bt = B^T row-major [Ncols,K].  Block = 256 threads = 8 wave32.
// TDM double-buffered staging: wave 0 issues next K-slab while all waves
// run WMMAs on the current one.  As/Bs are 2 x (128x32) halves each.
// ---------------------------------------------------------------------------
__device__ inline void gemm_tile(const half_t* __restrict__ A,
                                 const half_t* __restrict__ Bt,
                                 int K, int row0, int col0,
                                 half_t* As, half_t* Bs, v8f acc[4][2]) {
  const int tid  = threadIdx.x;
  const int lane = tid & 31;
  const int wave = tid >> 5;
  const int wm   = (wave >> 2) * 64;  // 0 or 64
  const int wn   = (wave & 3) * 32;   // 0..96
  const int ln   = lane & 15;
  const int g    = lane >> 4;
  const bool issuer = (wave == 0);

#pragma unroll
  for (int mt = 0; mt < 4; ++mt)
#pragma unroll
    for (int nt = 0; nt < 2; ++nt)
      acc[mt][nt] = (v8f)0.0f;

  const uint32_t asb = lds_lo32(As);
  const uint32_t bsb = lds_lo32(Bs);
  if (issuer) {
    tdm_load_2d(asb, A + (size_t)row0 * K, 32, 128, (uint32_t)K);
    tdm_load_2d(bsb, Bt + (size_t)col0 * K, 32, 128, (uint32_t)K);
  }

  int buf = 0;
  for (int kt = 0; kt < K; kt += 32, buf ^= 1) {
    __builtin_amdgcn_s_wait_tensorcnt(0);
    __syncthreads();
    if (issuer && (kt + 32 < K)) {
      // Prefetch next K-slab into the other buffer; DMA overlaps the WMMAs.
      tdm_load_2d(asb + (uint32_t)(buf ^ 1) * 8192u,
                  A + (size_t)row0 * K + kt + 32, 32, 128, (uint32_t)K);
      tdm_load_2d(bsb + (uint32_t)(buf ^ 1) * 8192u,
                  Bt + (size_t)col0 * K + kt + 32, 32, 128, (uint32_t)K);
    }
    const half_t* as = As + buf * 4096;
    const half_t* bs = Bs + buf * 4096;

    v16h af[4], bf[2];
#pragma unroll
    for (int mt = 0; mt < 4; ++mt) {
      // A operand 16x32: lanes 0-15 hold K {0..7,16..23}, lanes 16-31 +8.
      const half_t* p = as + (wm + mt * 16 + ln) * 32;
      af[mt] = cat8(*(const v8h*)(p + g * 8), *(const v8h*)(p + 16 + g * 8));
    }
#pragma unroll
    for (int nt = 0; nt < 2; ++nt) {
      // B operand 32x16: lanes 0-15 hold K {0..7,16..23}, lanes 16-31 +8.
      const half_t* p = bs + (wn + nt * 16 + ln) * 32 + g * 8;
      bf[nt] = cat8(*(const v8h*)p, *(const v8h*)(p + 16));
    }
#pragma unroll
    for (int mt = 0; mt < 4; ++mt)
#pragma unroll
      for (int nt = 0; nt < 2; ++nt)
        acc[mt][nt] = WMMA_F32_F16(af[mt], bf[nt], acc[mt][nt]);
  }
}

// ---------------------------------------------------------------------------
// Kernel 1: qkv = x @ w_qkv, scattering q,k as [bh][n][d] and v transposed
// as [bh][d][n] (all f16) so attention B-operands are contiguous.
// ---------------------------------------------------------------------------
__global__ __launch_bounds__(256) void gemm_qkv_kernel(const half_t* __restrict__ Xh,
                                                       const half_t* __restrict__ Wt,
                                                       half_t* __restrict__ qws,
                                                       half_t* __restrict__ kws,
                                                       half_t* __restrict__ vtws) {
  __shared__ __align__(16) half_t As[2 * 128 * 32];
  __shared__ __align__(16) half_t Bs[2 * 128 * 32];
  const int col0 = blockIdx.x * 128;   // 0..3071
  const int row0 = blockIdx.y * 128;   // 0..4095
  v8f acc[4][2];
  gemm_tile(Xh, Wt, 1024, row0, col0, As, Bs, acc);

  const int tid = threadIdx.x, lane = tid & 31, wave = tid >> 5;
  const int wm = (wave >> 2) * 64, wn = (wave & 3) * 32;
  const int ln = lane & 15, g = lane >> 4;
#pragma unroll
  for (int mt = 0; mt < 4; ++mt)
#pragma unroll
    for (int nt = 0; nt < 2; ++nt)
#pragma unroll
      for (int r = 0; r < 8; ++r) {
        int row = row0 + wm + mt * 16 + r + 8 * g;  // global M index
        int col = col0 + wn + nt * 16 + ln;         // global 3C index
        int b = row >> 11, n = row & 2047;
        int which = col >> 10, rem = col & 1023;
        int h = rem >> 6, d = rem & 63;
        int bh = b * 16 + h;
        half_t v = (half_t)acc[mt][nt][r];
        if (which == 0)      qws [(size_t)(bh * 2048 + n) * 64 + d]  = v;
        else if (which == 1) kws [(size_t)(bh * 2048 + n) * 64 + d]  = v;
        else                 vtws[(size_t)(bh * 64 + d) * 2048 + n]  = v;
      }
}

// ---------------------------------------------------------------------------
// Kernel 2: flash attention. Block = 128 threads (4 waves) handles 64 q rows
// of one (b,h); each wave owns a 16-row q tile with q fragments in registers.
// K/V blocks staged by TDM (double buffered); online softmax in f32.
// ---------------------------------------------------------------------------
__global__ __launch_bounds__(128) void flash_attn_kernel(const half_t* __restrict__ qws,
                                                         const half_t* __restrict__ kws,
                                                         const half_t* __restrict__ vtws,
                                                         half_t* __restrict__ attn_ws) {
  __shared__ __align__(16) half_t Ks[2 * 64 * 64];   // k rows [j][d], x2 buffers
  __shared__ __align__(16) half_t Vs[2 * 64 * 64];   // v^T rows [d][j], x2 buffers
  __shared__ __align__(16) half_t Ps[4 * 16 * 64];   // per-wave P staging
  const int bh = blockIdx.y;           // 0..31
  const int q0 = blockIdx.x * 64;      // q row block
  const int tid = threadIdx.x, lane = tid & 31, wave = tid >> 5;
  const int ln = lane & 15, g = lane >> 4;
  const int b = bh >> 4, h = bh & 15;
  const bool issuer = (wave == 0);

  const half_t* qbase = qws  + (size_t)bh * 2048 * 64;
  const half_t* kbase = kws  + (size_t)bh * 2048 * 64;
  const half_t* vbase = vtws + (size_t)bh * 64 * 2048;

  // Persistent q fragments: 16x64 split into two 16x32 A-operands (d halves).
  const int qrow = q0 + wave * 16 + ln;
  v16h qf[2];
#pragma unroll
  for (int kk = 0; kk < 2; ++kk) {
    const half_t* p = qbase + (size_t)qrow * 64 + kk * 32 + g * 8;
    qf[kk] = cat8(*(const v8h*)p, *(const v8h*)(p + 16));
  }

  v8f oacc[4];
#pragma unroll
  for (int dt = 0; dt < 4; ++dt) oacc[dt] = (v8f)0.0f;
  float mst[8], lst[8];
#pragma unroll
  for (int r = 0; r < 8; ++r) { mst[r] = -1e30f; lst[r] = 0.0f; }

  const float scale = 0.125f;  // Dh^-0.5, Dh = 64
  const uint32_t ksb = lds_lo32(Ks);
  const uint32_t vsb = lds_lo32(Vs);

  if (issuer) {
    // K rows are contiguous (64 rows x 64 halves): one flat 1D tile.
    tdm_load_2d(ksb, kbase, 4096, 1, 4096);
    // V^T block: 64 d-rows of 64 halves, row stride N=2048 halves.
    tdm_load_2d(vsb, vbase, 64, 64, 2048);
  }

  int buf = 0;
  for (int j0 = 0; j0 < 2048; j0 += 64, buf ^= 1) {
    __builtin_amdgcn_s_wait_tensorcnt(0);
    __syncthreads();
    if (issuer && (j0 + 64 < 2048)) {
      tdm_load_2d(ksb + (uint32_t)(buf ^ 1) * 8192u,
                  kbase + (size_t)(j0 + 64) * 64, 4096, 1, 4096);
      tdm_load_2d(vsb + (uint32_t)(buf ^ 1) * 8192u,
                  vbase + (j0 + 64), 64, 64, 2048);
    }
    const half_t* ks = Ks + buf * 4096;
    const half_t* vs = Vs + buf * 4096;

    // Scores: s[jt] = q(16x64) . k_tile^T(64x16), two K=32 WMMA steps each.
    v8f s[4];
#pragma unroll
    for (int jt = 0; jt < 4; ++jt) {
      v8f z = (v8f)0.0f;
#pragma unroll
      for (int kk = 0; kk < 2; ++kk) {
        const half_t* p = ks + (jt * 16 + ln) * 64 + kk * 32 + g * 8;
        v16h bf = cat8(*(const v8h*)p, *(const v8h*)(p + 16));
        z = WMMA_F32_F16(qf[kk], bf, z);
      }
#pragma unroll
      for (int r = 0; r < 8; ++r) z[r] *= scale;
      s[jt] = z;
    }

    // Row max across the 64-wide block (columns stripe over 16-lane halves).
    float rmax[8];
#pragma unroll
    for (int r = 0; r < 8; ++r) {
      float mx = fmaxf(fmaxf(s[0][r], s[1][r]), fmaxf(s[2][r], s[3][r]));
      mx = fmaxf(mx, __shfl_xor(mx, 1));
      mx = fmaxf(mx, __shfl_xor(mx, 2));
      mx = fmaxf(mx, __shfl_xor(mx, 4));
      mx = fmaxf(mx, __shfl_xor(mx, 8));
      rmax[r] = fmaxf(mst[r], mx);
    }

    // P = exp(s - rmax): accumulate row sums, stage P (f16) for A-operand.
    half_t* pw = Ps + wave * (16 * 64);
    float rsum[8];
#pragma unroll
    for (int r = 0; r < 8; ++r) rsum[r] = 0.0f;
#pragma unroll
    for (int jt = 0; jt < 4; ++jt)
#pragma unroll
      for (int r = 0; r < 8; ++r) {
        float p = __expf(s[jt][r] - rmax[r]);
        rsum[r] += p;
        pw[(r + 8 * g) * 64 + jt * 16 + ln] = (half_t)p;
      }

    // Update running stats, rescale output accumulators.
#pragma unroll
    for (int r = 0; r < 8; ++r) {
      float su = rsum[r];
      su += __shfl_xor(su, 1);
      su += __shfl_xor(su, 2);
      su += __shfl_xor(su, 4);
      su += __shfl_xor(su, 8);
      float alpha = __expf(mst[r] - rmax[r]);
      lst[r] = lst[r] * alpha + su;
      mst[r] = rmax[r];
#pragma unroll
      for (int dt = 0; dt < 4; ++dt) oacc[dt][r] *= alpha;
    }

    // oacc += P(16x64) . V(64x64): A from Ps, B from Vs (v^T rows contiguous).
#pragma unroll
    for (int kk = 0; kk < 2; ++kk) {
      const half_t* pa = pw + ln * 64 + kk * 32 + g * 8;
      v16h pf = cat8(*(const v8h*)pa, *(const v8h*)(pa + 16));
#pragma unroll
      for (int dt = 0; dt < 4; ++dt) {
        const half_t* pb = vs + (dt * 16 + ln) * 64 + kk * 32 + g * 8;
        v16h vf = cat8(*(const v8h*)pb, *(const v8h*)(pb + 16));
        oacc[dt] = WMMA_F32_F16(pf, vf, oacc[dt]);
      }
    }
  }

  // Normalize and write f16 in [B][N][H*Dh] layout for the proj GEMM.
#pragma unroll
  for (int dt = 0; dt < 4; ++dt)
#pragma unroll
    for (int r = 0; r < 8; ++r) {
      int row = q0 + wave * 16 + r + 8 * g;
      float val = oacc[dt][r] / lst[r];
      attn_ws[(size_t)(b * 2048 + row) * 1024 + h * 64 + dt * 16 + ln] = (half_t)val;
    }
}

// ---------------------------------------------------------------------------
// Kernel 3: out = attn @ w_proj + bias (f32 output).
// ---------------------------------------------------------------------------
__global__ __launch_bounds__(256) void gemm_proj_kernel(const half_t* __restrict__ Ah,
                                                        const half_t* __restrict__ Wt,
                                                        const float* __restrict__ bias,
                                                        float* __restrict__ out) {
  __shared__ __align__(16) half_t As[2 * 128 * 32];
  __shared__ __align__(16) half_t Bs[2 * 128 * 32];
  const int col0 = blockIdx.x * 128;
  const int row0 = blockIdx.y * 128;
  v8f acc[4][2];
  gemm_tile(Ah, Wt, 1024, row0, col0, As, Bs, acc);

  const int tid = threadIdx.x, lane = tid & 31, wave = tid >> 5;
  const int wm = (wave >> 2) * 64, wn = (wave & 3) * 32;
  const int ln = lane & 15, g = lane >> 4;
#pragma unroll
  for (int mt = 0; mt < 4; ++mt)
#pragma unroll
    for (int nt = 0; nt < 2; ++nt) {
      int col = col0 + wn + nt * 16 + ln;
      float bv = bias[col];
#pragma unroll
      for (int r = 0; r < 8; ++r) {
        int row = row0 + wm + mt * 16 + r + 8 * g;
        out[(size_t)row * 1024 + col] = acc[mt][nt][r] + bv;
      }
    }
}

// ---------------------------------------------------------------------------
// Host launcher
// ---------------------------------------------------------------------------
extern "C" void kernel_launch(void* const* d_in, const int* in_sizes, int n_in,
                              void* d_out, int out_size, void* d_ws, size_t ws_size,
                              hipStream_t stream) {
  const float* x      = (const float*)d_in[0];
  const float* w_qkv  = (const float*)d_in[1];
  const float* w_proj = (const float*)d_in[2];
  const float* b_proj = (const float*)d_in[3];
  float* out = (float*)d_out;
  (void)in_sizes; (void)n_in; (void)out_size; (void)ws_size;

  const int B = 2, N = 2048, C = 1024, H = 16, Dh = 64;
  const size_t M = (size_t)B * N;  // 4096

  char* ws = (char*)d_ws;
  size_t off = 0;
  auto alloc = [&](size_t bytes) -> char* {
    char* p = ws + off;
    off += (bytes + 255) & ~(size_t)255;
    return p;
  };
  half_t* x_h   = (half_t*)alloc(M * C * 2);                      // 8 MB
  half_t* wqkvt = (half_t*)alloc((size_t)3 * C * C * 2);          // 6 MB (w_qkv^T)
  half_t* wprjt = (half_t*)alloc((size_t)C * C * 2);              // 2 MB (w_proj^T)
  half_t* q_ws  = (half_t*)alloc((size_t)B * H * N * Dh * 2);     // 8 MB
  half_t* k_ws  = (half_t*)alloc((size_t)B * H * N * Dh * 2);     // 8 MB
  half_t* vt_ws = (half_t*)alloc((size_t)B * H * Dh * N * 2);     // 8 MB (v transposed)
  half_t* a_ws  = (half_t*)alloc(M * C * 2);                      // 8 MB

  {
    int n = (int)(M * C);
    cvt_f32_to_f16<<<(n + 255) / 256, 256, 0, stream>>>(x, x_h, n);
  }
  {
    int n = 3 * C * C;
    transpose_f32_to_f16<<<(n + 255) / 256, 256, 0, stream>>>(w_qkv, wqkvt, C, 3 * C);
  }
  {
    int n = C * C;
    transpose_f32_to_f16<<<(n + 255) / 256, 256, 0, stream>>>(w_proj, wprjt, C, C);
  }

  gemm_qkv_kernel<<<dim3(3 * C / 128, (int)(M / 128)), 256, 0, stream>>>(
      x_h, wqkvt, q_ws, k_ws, vt_ws);

  flash_attn_kernel<<<dim3(N / 64, B * H), 128, 0, stream>>>(
      q_ws, k_ws, vt_ws, a_ws);

  gemm_proj_kernel<<<dim3(C / 128, (int)(M / 128)), 256, 0, stream>>>(
      a_ws, wprjt, b_proj, out);
}
